// EBTBlock_48773648614148
// MI455X (gfx1250) — matrix operand order repaired and checked by reading, not compiled, not run on hardware
//
#include <hip/hip_runtime.h>

typedef __bf16 bf16_t;
typedef __attribute__((ext_vector_type(16))) __bf16 v16bf;
typedef __attribute__((ext_vector_type(8))) __bf16 v8bf;
typedef __attribute__((ext_vector_type(8))) float v8f;

static __device__ __forceinline__ v8f wmma_bf16(v16bf a, v16bf b, v8f c) {
  // D = A(16x32 bf16) * B(32x16 bf16) + C(16x16 f32)
  return __builtin_amdgcn_wmma_f32_16x16x32_bf16(
      /*neg_a=*/false, a, /*neg_b=*/false, b,
      /*c_mod=*/(short)0, c, /*reuse_a=*/false, /*reuse_b=*/false);
}

static __device__ __forceinline__ v16bf cat8(v8bf lo, v8bf hi) {
  v16bf r;
#pragma unroll
  for (int i = 0; i < 8; ++i) {
    r[i] = lo[i];
    r[i + 8] = hi[i];
  }
  return r;
}

// A-fragment pack permutation within a 32-K group: identity (contiguous k).
static __device__ __forceinline__ int packpos(int km) {
  return km;
}

// ---------------------------------------------------------------------------
// GEMM: C[M,N] = Apk[M,K] * Bbf[N,K]^T
// Apk: packed bf16 activations (32-K groups permuted so a lane's A-fragment is
//      one contiguous 32B chunk at row*K + k0 + 16*hi).
// Bbf: plain bf16 weights [N,K] (B-fragment contiguous by construction).
// Block = 256 threads = 8 waves (4M x 2N); wave tile = 32M x 64N:
// each B fragment feeds two A fragments -> 8 WMMAs per 6 fragment loads.
// Explicit ping-pong double buffering: zero converts, zero rotation copies.
// ---------------------------------------------------------------------------
__global__ __launch_bounds__(256) void gemm_bf16_wmma_kernel(
    const bf16_t* __restrict__ Apk, const bf16_t* __restrict__ Bbf,
    float* __restrict__ C, int M, int N, int K) {
  const int lane = threadIdx.x & 31;
  const int wave = threadIdx.x >> 5;
  const int lr = lane & 15;
  const int hi = lane >> 4;
  const int Mbase = blockIdx.y * 128 + (wave >> 1) * 32;
  const int Nbase = blockIdx.x * 128 + (wave & 1) * 64;

  const v8f vzero = {0.f, 0.f, 0.f, 0.f, 0.f, 0.f, 0.f, 0.f};
  v8f acc[2][4];
#pragma unroll
  for (int mi = 0; mi < 2; ++mi)
#pragma unroll
    for (int j = 0; j < 4; ++j) acc[mi][j] = vzero;

  const bf16_t* arow0 = Apk + (size_t)(Mbase + lr) * K + 16 * hi;
  const bf16_t* arow1 = arow0 + (size_t)16 * K;
  const bf16_t* br0 = Bbf + (size_t)(Nbase + lr) * K + 16 * hi;
  const bf16_t* br1 = br0 + (size_t)16 * K;
  const bf16_t* br2 = br0 + (size_t)32 * K;
  const bf16_t* br3 = br0 + (size_t)48 * K;

  v16bf a00, a01, p00, p01, p02, p03;  // buffer 0
  v16bf a10, a11, p10, p11, p12, p13;  // buffer 1

#define LD0(k)                           \
  a00 = *(const v16bf*)(arow0 + (k));    \
  a01 = *(const v16bf*)(arow1 + (k));    \
  p00 = *(const v16bf*)(br0 + (k));      \
  p01 = *(const v16bf*)(br1 + (k));      \
  p02 = *(const v16bf*)(br2 + (k));      \
  p03 = *(const v16bf*)(br3 + (k));
#define LD1(k)                           \
  a10 = *(const v16bf*)(arow0 + (k));    \
  a11 = *(const v16bf*)(arow1 + (k));    \
  p10 = *(const v16bf*)(br0 + (k));      \
  p11 = *(const v16bf*)(br1 + (k));      \
  p12 = *(const v16bf*)(br2 + (k));      \
  p13 = *(const v16bf*)(br3 + (k));
#define MMA0()                                     \
  acc[0][0] = wmma_bf16(a00, p00, acc[0][0]);      \
  acc[1][0] = wmma_bf16(a01, p00, acc[1][0]);      \
  acc[0][1] = wmma_bf16(a00, p01, acc[0][1]);      \
  acc[1][1] = wmma_bf16(a01, p01, acc[1][1]);      \
  acc[0][2] = wmma_bf16(a00, p02, acc[0][2]);      \
  acc[1][2] = wmma_bf16(a01, p02, acc[1][2]);      \
  acc[0][3] = wmma_bf16(a00, p03, acc[0][3]);      \
  acc[1][3] = wmma_bf16(a01, p03, acc[1][3]);
#define MMA1()                                     \
  acc[0][0] = wmma_bf16(a10, p10, acc[0][0]);      \
  acc[1][0] = wmma_bf16(a11, p10, acc[1][0]);      \
  acc[0][1] = wmma_bf16(a10, p11, acc[0][1]);      \
  acc[1][1] = wmma_bf16(a11, p11, acc[1][1]);      \
  acc[0][2] = wmma_bf16(a10, p12, acc[0][2]);      \
  acc[1][2] = wmma_bf16(a11, p12, acc[1][2]);      \
  acc[0][3] = wmma_bf16(a10, p13, acc[0][3]);      \
  acc[1][3] = wmma_bf16(a11, p13, acc[1][3]);

  // K is a multiple of 64 for every GEMM in this block (768, 2048).
  LD0(0);
  LD1(32);
  int k0 = 0;
  for (; k0 + 64 < K; k0 += 64) {
    MMA0();
    LD0(k0 + 64);  // in flight under MMA1
    if (k0 + 128 < K) {
      __builtin_prefetch(arow0 + k0 + 128, 0, 1);  // global_prefetch_b8
      __builtin_prefetch(br0 + k0 + 128, 0, 1);
    }
    MMA1();
    LD1(k0 + 96);  // in flight under next MMA0
  }
  MMA0();
  MMA1();
#undef LD0
#undef LD1
#undef MMA0
#undef MMA1

  // C/D layout: vgpr r, lane l -> (M = r + 8*(l/16), N = l%16)
#pragma unroll
  for (int mi = 0; mi < 2; ++mi) {
#pragma unroll
    for (int j = 0; j < 4; ++j) {
      const int n = Nbase + 16 * j + lr;
#pragma unroll
      for (int r = 0; r < 8; ++r) {
        const int m = Mbase + 16 * mi + r + 8 * hi;
        C[(size_t)m * N + n] = acc[mi][j][r];
      }
    }
  }
}

// ---------------------------------------------------------------------------
// Plain f32 -> bf16 conversion (weights), vectorized x4. n % 4 == 0.
// ---------------------------------------------------------------------------
__global__ __launch_bounds__(256) void convert_bf16_kernel(
    const float* __restrict__ in, bf16_t* __restrict__ out, long long n4) {
  const long long i = (long long)blockIdx.x * 256 + threadIdx.x;
  if (i >= n4) return;
  const float4 f = ((const float4*)in)[i];
  bf16_t* o = out + i * 4;
  o[0] = (bf16_t)f.x;
  o[1] = (bf16_t)f.y;
  o[2] = (bf16_t)f.z;
  o[3] = (bf16_t)f.w;
}

// ---------------------------------------------------------------------------
// f32 -> packed bf16 A-layout: out[row*K + (k&~31) + packpos(k&31)] = in[row*K+k]
// ---------------------------------------------------------------------------
__global__ __launch_bounds__(256) void convert_pack_kernel(
    const float* __restrict__ in, bf16_t* __restrict__ out, long long rows,
    int K) {
  const long long total = rows * K;
  const long long idx = (long long)blockIdx.x * 256 + threadIdx.x;
  if (idx >= total) return;
  const int k = (int)(idx % K);
  const long long row = idx / K;
  out[row * K + (k & ~31) + packpos(k & 31)] = (bf16_t)in[idx];
}

// ---------------------------------------------------------------------------
// RoPE + split: qkv fp32 [B,S,3H,D] -> Qb/Kb (rope'd) bf16 [B,H,S,D],
//               Vt bf16 [B,H,D,S] (transposed: PV B-fragments contiguous)
// ---------------------------------------------------------------------------
__global__ __launch_bounds__(256) void rope_split_kernel(
    const float* __restrict__ qkv, const float* __restrict__ cosT,
    const float* __restrict__ sinT, bf16_t* __restrict__ Qb,
    bf16_t* __restrict__ Kb, bf16_t* __restrict__ Vt, int Bn, int S, int Hn) {
  const long long total = (long long)Bn * S * Hn * 64;
  long long idx = (long long)blockIdx.x * 256 + threadIdx.x;
  if (idx >= total) return;
  const int d = (int)(idx & 63);
  long long t = idx >> 6;
  const int h = (int)(t % Hn);
  t /= Hn;
  const int s = (int)(t % S);
  const int b = (int)(t / S);
  const size_t base = ((size_t)b * S + s) * (size_t)(3 * Hn * 64);
  const float q = qkv[base + h * 64 + d];
  const float k = qkv[base + (Hn + h) * 64 + d];
  const float v = qkv[base + (2 * Hn + h) * 64 + d];
  const int d2 = (d < 32) ? (d + 32) : (d - 32);
  const float sgn = (d < 32) ? -1.0f : 1.0f;
  const float qr = sgn * qkv[base + h * 64 + d2];
  const float kr = sgn * qkv[base + (Hn + h) * 64 + d2];
  const float c = cosT[s * 64 + d];
  const float sn = sinT[s * 64 + d];
  const size_t o = (((size_t)b * Hn + h) * S + s) * 64 + d;
  Qb[o] = (bf16_t)(q * c + qr * sn);
  Kb[o] = (bf16_t)(k * c + kr * sn);
  const size_t ot = (((size_t)b * Hn + h) * 64 + d) * S + s;
  Vt[ot] = (bf16_t)v;
}

// ---------------------------------------------------------------------------
// Flash attention: block = 4 waves, each wave owns 16 query rows of one (b,h).
// Key blocks of 32: 4 score WMMAs, online softmax, 4 PV WMMAs.
// Epilogue writes ctx as packed bf16 A-layout for the O-projection GEMM.
// ---------------------------------------------------------------------------
__global__ __launch_bounds__(128) void flash_attn_kernel(
    const bf16_t* __restrict__ Qb, const bf16_t* __restrict__ Kb,
    const bf16_t* __restrict__ Vt, bf16_t* __restrict__ ctx_pk, int Hn, int S,
    float scale) {
  const int bh = blockIdx.x;
  const int b = bh / Hn, h = bh % Hn;
  const int lane = threadIdx.x & 31;
  const int wave = threadIdx.x >> 5;
  const int lr = lane & 15;
  const int hi = lane >> 4;
  const int qbase = blockIdx.y * 64 + wave * 16;

  __shared__ __align__(16) bf16_t Plds[4][16][32];

  const bf16_t* Qp = Qb + (size_t)bh * S * 64;
  const bf16_t* Kp = Kb + (size_t)bh * S * 64;
  const bf16_t* Vp = Vt + (size_t)bh * 64 * S;

  v16bf qf[2];
  {
    const bf16_t* qrow = Qp + (size_t)(qbase + lr) * 64;
#pragma unroll
    for (int t = 0; t < 2; ++t) {
      qf[t] = *(const v16bf*)(qrow + 32 * t + 16 * hi);
    }
  }

  const v8f vzero = {0.f, 0.f, 0.f, 0.f, 0.f, 0.f, 0.f, 0.f};
  float mrow[8], lrow[8];
  v8f O[4];
#pragma unroll
  for (int r = 0; r < 8; ++r) {
    mrow[r] = -1e30f;
    lrow[r] = 0.f;
  }
#pragma unroll
  for (int j = 0; j < 4; ++j) O[j] = vzero;

  for (int kb = 0; kb < S; kb += 32) {
    v8f sacc[2];
    sacc[0] = vzero;
    sacc[1] = vzero;
#pragma unroll
    for (int j = 0; j < 2; ++j) {
      const bf16_t* krow = Kp + (size_t)(kb + 16 * j + lr) * 64;
#pragma unroll
      for (int t = 0; t < 2; ++t) {
        const v16bf kf = *(const v16bf*)(krow + 32 * t + 16 * hi);
        sacc[j] = wmma_bf16(qf[t], kf, sacc[j]);
      }
    }
    float p0[8], p1[8];
#pragma unroll
    for (int r = 0; r < 8; ++r) {
      const float s0 = sacc[0][r] * scale;
      const float s1 = sacc[1][r] * scale;
      float mx = fmaxf(s0, s1);
#pragma unroll
      for (int off = 1; off < 16; off <<= 1) mx = fmaxf(mx, __shfl_xor(mx, off));
      const float mnew = fmaxf(mrow[r], mx);
      const float alpha = __expf(mrow[r] - mnew);
      p0[r] = __expf(s0 - mnew);
      p1[r] = __expf(s1 - mnew);
      float ps = p0[r] + p1[r];
#pragma unroll
      for (int off = 1; off < 16; off <<= 1) ps += __shfl_xor(ps, off);
      lrow[r] = lrow[r] * alpha + ps;
      mrow[r] = mnew;
#pragma unroll
      for (int j = 0; j < 4; ++j) O[j][r] = O[j][r] * alpha;
    }
    __syncthreads();
#pragma unroll
    for (int r = 0; r < 8; ++r) {
      Plds[wave][r + 8 * hi][lr] = (bf16_t)p0[r];
      Plds[wave][r + 8 * hi][lr + 16] = (bf16_t)p1[r];
    }
    __syncthreads();
    const v8bf plo = *(const v8bf*)(&Plds[wave][lr][8 * hi]);
    const v8bf phi = *(const v8bf*)(&Plds[wave][lr][16 + 8 * hi]);
    const v16bf pa = cat8(plo, phi);
#pragma unroll
    for (int j = 0; j < 4; ++j) {
      const bf16_t* vtcol = Vp + (size_t)(16 * j + lr) * S;
      const v16bf vf = cat8(*(const v8bf*)(vtcol + kb + 8 * hi),
                            *(const v8bf*)(vtcol + kb + 16 + 8 * hi));
      O[j] = wmma_bf16(pa, vf, O[j]);
    }
  }
  // normalize and store ctx as packed bf16 [B*S, H*64] for the next GEMM
#pragma unroll
  for (int r = 0; r < 8; ++r) {
    const float inv = 1.0f / lrow[r];
    const int m = qbase + r + 8 * hi;
#pragma unroll
    for (int j = 0; j < 4; ++j) {
      const int kc = 16 * j + lr;  // 0..63 within this head's columns
      const int q = (kc & ~31) + packpos(kc & 31);
      ctx_pk[((size_t)b * S + m) * (Hn * 64) + h * 64 + q] =
          (bf16_t)(O[j][r] * inv);
    }
  }
}

// ---------------------------------------------------------------------------
// y = rms_norm(a + b); optional packed-bf16 copy for downstream GEMM A.
// ---------------------------------------------------------------------------
__global__ __launch_bounds__(256) void add_rmsnorm_kernel(
    const float* __restrict__ a, const float* __restrict__ b,
    float* __restrict__ y, bf16_t* __restrict__ ybf, int C, float eps) {
  const int row = blockIdx.x;
  const float* ar = a + (size_t)row * C;
  const float* br = b + (size_t)row * C;
  __shared__ float red[256];
  float ss = 0.f;
  for (int i = threadIdx.x; i < C; i += 256) {
    const float t = ar[i] + br[i];
    ss += t * t;
  }
  red[threadIdx.x] = ss;
  __syncthreads();
  for (int s = 128; s > 0; s >>= 1) {
    if (threadIdx.x < s) red[threadIdx.x] += red[threadIdx.x + s];
    __syncthreads();
  }
  const float inv = rsqrtf(red[0] / (float)C + eps);
  for (int i = threadIdx.x; i < C; i += 256) {
    const float val = (ar[i] + br[i]) * inv;
    y[(size_t)row * C + i] = val;
    if (ybf) ybf[(size_t)row * C + (i & ~31) + packpos(i & 31)] = (bf16_t)val;
  }
}

// ---------------------------------------------------------------------------
// out (packed bf16) = silu(gate) * up from gu fp32 [rows, 2*inter]
// ---------------------------------------------------------------------------
__global__ __launch_bounds__(256) void silu_mul_kernel(
    const float* __restrict__ gu, bf16_t* __restrict__ out, long long rows,
    int inter) {
  const long long total = rows * inter;
  long long idx = (long long)blockIdx.x * 256 + threadIdx.x;
  if (idx >= total) return;
  const long long row = idx / inter;
  const int i = (int)(idx % inter);
  const float g = gu[row * (2LL * inter) + i];
  const float u = gu[row * (2LL * inter) + inter + i];
  const float val = (g / (1.f + __expf(-g))) * u;
  out[row * inter + (i & ~31) + packpos(i & 31)] = (bf16_t)val;
}

// ---------------------------------------------------------------------------
extern "C" void kernel_launch(void* const* d_in, const int* in_sizes, int n_in,
                              void* d_out, int out_size, void* d_ws,
                              size_t ws_size, hipStream_t stream) {
  const float* hs = (const float*)d_in[0];
  const float* cosT = (const float*)d_in[1];
  const float* sinT = (const float*)d_in[2];
  const float* w_qkv = (const float*)d_in[3];
  const float* w_o = (const float*)d_in[4];
  const float* w_gu = (const float*)d_in[5];
  const float* w_down = (const float*)d_in[6];
  float* out = (float*)d_out;

  const int B = 2, S = 2048, C = 768, H = 12, D = 64, I = 2048;
  const int M = B * S;  // 4096 rows

  char* ws = (char*)d_ws;
  size_t off = 0;
  auto alloc = [&](size_t bytes) -> void* {
    void* p = (void*)(ws + off);
    off = (off + bytes + 255) & ~(size_t)255;
    return p;
  };
  float* qkv = (float*)alloc((size_t)M * 3 * H * D * sizeof(float));
  bf16_t* hs_pk = (bf16_t*)alloc((size_t)M * C * sizeof(bf16_t));
  bf16_t* wqkv_bf = (bf16_t*)alloc((size_t)3 * H * D * C * sizeof(bf16_t));
  bf16_t* wo_bf = (bf16_t*)alloc((size_t)C * H * D * sizeof(bf16_t));
  bf16_t* wgu_bf = (bf16_t*)alloc((size_t)2 * I * C * sizeof(bf16_t));
  bf16_t* wdn_bf = (bf16_t*)alloc((size_t)C * I * sizeof(bf16_t));
  bf16_t* Qb = (bf16_t*)alloc((size_t)M * H * D * sizeof(bf16_t));
  bf16_t* Kb = (bf16_t*)alloc((size_t)M * H * D * sizeof(bf16_t));
  bf16_t* Vt = (bf16_t*)alloc((size_t)M * H * D * sizeof(bf16_t));
  bf16_t* ctx_pk = (bf16_t*)alloc((size_t)M * H * D * sizeof(bf16_t));
  float* attn_out = (float*)alloc((size_t)M * C * sizeof(float));
  float* x = (float*)alloc((size_t)M * C * sizeof(float));
  bf16_t* x_pk = (bf16_t*)alloc((size_t)M * C * sizeof(bf16_t));
  float* gu = (float*)alloc((size_t)M * 2 * I * sizeof(float));
  bf16_t* hmid_pk = (bf16_t*)alloc((size_t)M * I * sizeof(bf16_t));
  float* mlp = (float*)alloc((size_t)M * C * sizeof(float));

  const dim3 blk256(256);
  auto grid1d = [](long long n) { return dim3((unsigned)((n + 255) / 256)); };

  // 0. one-time conversions: weights -> bf16, hidden_states -> packed bf16
  convert_bf16_kernel<<<grid1d((long long)3 * H * D * C / 4), blk256, 0,
                        stream>>>(w_qkv, wqkv_bf, (long long)3 * H * D * C / 4);
  convert_bf16_kernel<<<grid1d((long long)C * H * D / 4), blk256, 0, stream>>>(
      w_o, wo_bf, (long long)C * H * D / 4);
  convert_bf16_kernel<<<grid1d((long long)2 * I * C / 4), blk256, 0, stream>>>(
      w_gu, wgu_bf, (long long)2 * I * C / 4);
  convert_bf16_kernel<<<grid1d((long long)C * I / 4), blk256, 0, stream>>>(
      w_down, wdn_bf, (long long)C * I / 4);
  convert_pack_kernel<<<grid1d((long long)M * C), blk256, 0, stream>>>(
      hs, hs_pk, (long long)M, C);

  // 1. qkv = hs @ w_qkv^T   [4096 x 2304]
  gemm_bf16_wmma_kernel<<<dim3((3 * H * D) / 128, M / 128), blk256, 0,
                          stream>>>(hs_pk, wqkv_bf, qkv, M, 3 * H * D, C);

  // 2. RoPE + head split -> Q/K [B,H,S,D], Vt [B,H,D,S] (bf16)
  rope_split_kernel<<<grid1d((long long)B * S * H * D), blk256, 0, stream>>>(
      qkv, cosT, sinT, Qb, Kb, Vt, B, S, H);

  // 3. flash attention -> ctx packed bf16 [B*S, H*64]
  flash_attn_kernel<<<dim3(B * H, S / 64), dim3(128), 0, stream>>>(
      Qb, Kb, Vt, ctx_pk, H, S, 0.125f);

  // 4. attn_out = ctx @ w_o^T   [4096 x 768]
  gemm_bf16_wmma_kernel<<<dim3(C / 128, M / 128), blk256, 0, stream>>>(
      ctx_pk, wo_bf, attn_out, M, C, H * D);

  // 5. x = rms_norm(hs + attn_out)  (+ packed bf16 copy)
  add_rmsnorm_kernel<<<dim3(M), blk256, 0, stream>>>(hs, attn_out, x, x_pk, C,
                                                     1e-5f);

  // 6. gu = x @ w_gate_up^T   [4096 x 4096]
  gemm_bf16_wmma_kernel<<<dim3((2 * I) / 128, M / 128), blk256, 0, stream>>>(
      x_pk, wgu_bf, gu, M, 2 * I, C);

  // 7. hmid = silu(gate) * up  (packed bf16)
  silu_mul_kernel<<<grid1d((long long)M * I), blk256, 0, stream>>>(
      gu, hmid_pk, (long long)M, I);

  // 8. mlp = hmid @ w_down^T   [4096 x 768]
  gemm_bf16_wmma_kernel<<<dim3(C / 128, M / 128), blk256, 0, stream>>>(
      hmid_pk, wdn_bf, mlp, M, C, I);

  // 9. out = rms_norm(x + mlp)
  add_rmsnorm_kernel<<<dim3(M), blk256, 0, stream>>>(x, mlp, out, nullptr, C,
                                                     1e-5f);
}
